// RealNVP_66907000537609
// MI455X (gfx1250) — compile-verified, matrix operand order checked by reading, not run on hardware
//
#include <hip/hip_runtime.h>
#include <stdint.h>

#define BATCH   32768
#define FDIM    784
#define UDIM    1024
#define NFLOWS  5
#define EPSV    1e-5f
#define LOG2PI  1.8378770664093453f

#define TM   128   // block M tile
#define TK   32    // k step (one bf16 WMMA)
#define LDT  40    // LDS row stride (bf16 elems), padded vs 32 to break bank conflicts

typedef __attribute__((ext_vector_type(16))) __bf16 v16bf;
typedef __attribute__((ext_vector_type(8)))  float  v8f;

union Frag { v16bf v; uint4 q[2]; };

typedef __attribute__((address_space(3))) unsigned short lds_us;

// LDS byte-offset of a pointer into a __shared__ array (ptrtoint of AS(3) pointer)
__device__ __forceinline__ unsigned ldsoff(const unsigned short* p) {
    return (unsigned)(unsigned long long)(lds_us*)p;
}

// CDNA5 async L2->LDS copy, 16B per lane, tracked by ASYNCcnt (CDNA5 ISA)
__device__ __forceinline__ void async_b128(unsigned lds_addr, const void* gaddr) {
    asm volatile("global_load_async_to_lds_b128 %0, %1, off"
                 :: "v"(lds_addr), "v"(gaddr) : "memory");
}
// pipeline waits: allow the next tile's 4 copies to stay in flight / drain fully
__device__ __forceinline__ void wait_async_le4() {
    asm volatile("s_wait_asynccnt 0x4" ::: "memory");
}
__device__ __forceinline__ void wait_async_le0() {
    asm volatile("s_wait_asynccnt 0x0" ::: "memory");
}

__device__ __forceinline__ unsigned short f2bf(float f) {
    union { float f; unsigned int u; } x; x.f = f;
    unsigned int r = x.u + 0x7FFFu + ((x.u >> 16) & 1u);  // round-to-nearest-even
    return (unsigned short)(r >> 16);
}

// ---------------------------------------------------------------- tiny kernels
__global__ void zero_kernel(float* p, int n) {
    int i = blockIdx.x * blockDim.x + threadIdx.x;
    if (i < n) p[i] = 0.f;
}

// cur_x = src ; mx = bf16(src * mask)
__global__ void prep_kernel(const float* __restrict__ src, const int* __restrict__ mask,
                            float* __restrict__ curx, unsigned short* __restrict__ mx) {
    int idx = blockIdx.x * blockDim.x + threadIdx.x;
    if (idx >= BATCH * FDIM) return;
    int f = idx % FDIM;
    float v = src[idx];
    curx[idx] = v;
    mx[idx] = f2bf(v * (float)mask[f]);
}

// out[n*K + k] = bf16(in[k*N + n])   (convert + transpose to N-major)
__global__ void convT_kernel(const float* __restrict__ in, unsigned short* __restrict__ out,
                             int K, int N) {
    int idx = blockIdx.x * blockDim.x + threadIdx.x;
    if (idx >= K * N) return;
    int k = idx / N, n = idx - k * N;
    out[(size_t)n * K + k] = f2bf(in[idx]);
}

// scalar: sum over flows/features of (log_gamma - 0.5*log(rv+eps))
__global__ void bnc_kernel(const float* __restrict__ lg, const float* __restrict__ rv,
                           float* __restrict__ bnc) {
    __shared__ float red[256];
    int tid = threadIdx.x;
    float s = 0.f;
    for (int i = tid; i < NFLOWS * FDIM; i += 256)
        s += lg[i] - 0.5f * __logf(rv[i] + EPSV);
    red[tid] = s; __syncthreads();
    for (int st = 128; st > 0; st >>= 1) {
        if (tid < st) red[tid] += red[tid + st];
        __syncthreads();
    }
    if (tid == 0) bnc[0] = red[0];
}

// ---------------------------------------------------------------- GEMM1: h = relu(mx @ W1 + b1)
// A: [BATCH x FDIM] bf16 row-major, BT: [UDIM x FDIM] bf16 (N-major), H: [BATCH x UDIM] bf16
__device__ __forceinline__ void g1_issue(const unsigned short* A, const unsigned short* BT,
                                         unsigned short* AsB, unsigned short* BsB,
                                         int m0, int n0, int K, int kb, int tid) {
    if (kb + TK <= K) {
        #pragma unroll
        for (int rep = 0; rep < 2; ++rep) {           // A tile: 128x32, async
            int u = tid + rep * 256, r = u >> 2, seg = u & 3;
            async_b128(ldsoff(&AsB[r * LDT + seg * 8]),
                       A + (size_t)(m0 + r) * K + kb + seg * 8);
        }
        #pragma unroll
        for (int rep = 0; rep < 2; ++rep) {           // B tile: 128 cols x 32 k, async
            int u = tid + rep * 256, r = u >> 2, seg = u & 3;
            async_b128(ldsoff(&BsB[r * LDT + seg * 8]),
                       BT + (size_t)(n0 + r) * K + kb + seg * 8);
        }
    } else {
        // K remainder (784 % 32 == 16): guarded VGPR path with zero-fill (DScnt path)
        #pragma unroll
        for (int rep = 0; rep < 2; ++rep) {
            int u = tid + rep * 256, r = u >> 2, seg = u & 3;
            int kk = kb + seg * 8;
            uint4 val = {0, 0, 0, 0};
            if (kk + 8 <= K) val = *(const uint4*)(A + (size_t)(m0 + r) * K + kk);
            *(uint4*)(&AsB[r * LDT + seg * 8]) = val;
        }
        #pragma unroll
        for (int rep = 0; rep < 2; ++rep) {
            int u = tid + rep * 256, r = u >> 2, seg = u & 3;
            int kk = kb + seg * 8;
            uint4 val = {0, 0, 0, 0};
            if (kk + 8 <= K) val = *(const uint4*)(BT + (size_t)(n0 + r) * K + kk);
            *(uint4*)(&BsB[r * LDT + seg * 8]) = val;
        }
    }
}

__global__ __launch_bounds__(256)
void gemm1_kernel(const unsigned short* __restrict__ A, const unsigned short* __restrict__ BT,
                  const float* __restrict__ bias, unsigned short* __restrict__ H) {
    __shared__ unsigned short As[2][TM * LDT];        // double-buffered
    __shared__ unsigned short Bs[2][128 * LDT];
    const int K = FDIM;
    const int KT = (K + TK - 1) / TK;                 // 25
    const int tid  = threadIdx.x;
    const int wave = tid >> 5, lane = tid & 31, lo = lane & 15, hi = lane >> 4;
    const int wm = wave >> 1, wn = wave & 1;          // 4x2 waves -> 32x64 wave tile
    const int m0 = blockIdx.y * TM;
    const int n0 = blockIdx.x * 128;

    v8f acc[2][4];
    #pragma unroll
    for (int i = 0; i < 2; ++i)
        #pragma unroll
        for (int j = 0; j < 4; ++j) { v8f z = {0,0,0,0,0,0,0,0}; acc[i][j] = z; }

    // prologue: prefetch tiles 0 and 1
    g1_issue(A, BT, As[0], Bs[0], m0, n0, K, 0, tid);
    g1_issue(A, BT, As[1], Bs[1], m0, n0, K, TK, tid);

    for (int kt = 0; kt < KT; ++kt) {
        const int p = kt & 1;
        // tile kt done when only tile kt+1's 4 async copies may remain outstanding
        if (kt + 1 < KT && (kt + 1) * TK + TK <= K) wait_async_le4();
        else                                        wait_async_le0();
        __syncthreads();

        Frag a[2], b[4];
        #pragma unroll
        for (int mi = 0; mi < 2; ++mi) {              // 16-bit A layout: K pairs, half split
            const unsigned short* pa = &As[p][(wm * 32 + mi * 16 + lo) * LDT];
            a[mi].q[0] = *(const uint4*)(pa + hi * 8);
            a[mi].q[1] = *(const uint4*)(pa + 16 + hi * 8);
        }
        #pragma unroll
        for (int ni = 0; ni < 4; ++ni) {
            const unsigned short* pb = &Bs[p][(wn * 64 + ni * 16 + lo) * LDT];
            b[ni].q[0] = *(const uint4*)(pb + hi * 8);
            b[ni].q[1] = *(const uint4*)(pb + 16 + hi * 8);
        }
        #pragma unroll
        for (int mi = 0; mi < 2; ++mi)
            #pragma unroll
            for (int ni = 0; ni < 4; ++ni)
                acc[mi][ni] = __builtin_amdgcn_wmma_f32_16x16x32_bf16(
                    false, a[mi].v, false, b[ni].v, (short)0, acc[mi][ni], false, false);
        __syncthreads();                              // all waves done reading buf p

        if (kt + 2 < KT)                              // overlap tile kt+2 with compute kt+1
            g1_issue(A, BT, As[p], Bs[p], m0, n0, K, (kt + 2) * TK, tid);
    }

    #pragma unroll
    for (int mi = 0; mi < 2; ++mi)
        #pragma unroll
        for (int ni = 0; ni < 4; ++ni) {
            int c  = n0 + wn * 64 + ni * 16 + lo;
            float bv = bias[c];
            #pragma unroll
            for (int v = 0; v < 8; ++v) {
                int r = m0 + wm * 32 + mi * 16 + v + hi * 8;   // C layout: row = v + 8*hi
                float x = acc[mi][ni][v] + bv;
                H[(size_t)r * UDIM + c] = f2bf(x > 0.f ? x : 0.f);
            }
        }
}

// ------------------------------------------ GEMM2 + coupling + batchnorm + ILDJ (fused epilogue)
// H: [BATCH x UDIM] bf16, W2T: [2*FDIM x UDIM] bf16 (N-major). t-cols = rows j, s-cols = rows F+j.
// N-overhang (784 % 64): B rows are clamped; garbage columns are discarded by the j<FDIM guard.
__device__ __forceinline__ void g2_issue(const unsigned short* Hm, const unsigned short* W2T,
                                         unsigned short* AsB, unsigned short* BtB,
                                         unsigned short* BsB,
                                         int m0, int kb, int jc, int tid, int lr, int lseg) {
    #pragma unroll
    for (int rep = 0; rep < 2; ++rep) {               // A tile (h) 128x32, async
        int u = tid + rep * 256, r = u >> 2, seg = u & 3;
        async_b128(ldsoff(&AsB[r * LDT + seg * 8]),
                   Hm + (size_t)(m0 + r) * UDIM + kb + seg * 8);
    }
    // Bt / Bs tiles: 64 cols x 32 k, async (clamped rows)
    async_b128(ldsoff(&BtB[lr * LDT + lseg * 8]),
               W2T + (size_t)jc * UDIM + kb + lseg * 8);
    async_b128(ldsoff(&BsB[lr * LDT + lseg * 8]),
               W2T + (size_t)(FDIM + jc) * UDIM + kb + lseg * 8);
}

__global__ __launch_bounds__(256)
void gemm2_kernel(const unsigned short* __restrict__ H, const unsigned short* __restrict__ W2T,
                  const float* __restrict__ b2, const int* __restrict__ mask,
                  const float* __restrict__ lg, const float* __restrict__ bt,
                  const float* __restrict__ rm, const float* __restrict__ rv,
                  float* __restrict__ curx, float* __restrict__ ildj) {
    __shared__ unsigned short As [2][TM * LDT];       // double-buffered
    __shared__ unsigned short Bts[2][64 * LDT];
    __shared__ unsigned short Bss[2][64 * LDT];
    const int KT = UDIM / TK;                         // 32 full tiles
    const int tid  = threadIdx.x;
    const int wave = tid >> 5, lane = tid & 31, lo = lane & 15, hi = lane >> 4;
    const int wm = wave >> 1, wn = wave & 1;          // 4x2 waves -> 32x32 wave tile
    const int m0 = blockIdx.y * TM;
    const int n0 = blockIdx.x * 64;

    v8f acct[2][2], accs[2][2];
    #pragma unroll
    for (int i = 0; i < 2; ++i)
        #pragma unroll
        for (int j = 0; j < 2; ++j) {
            v8f z = {0,0,0,0,0,0,0,0}; acct[i][j] = z; accs[i][j] = z;
        }

    const int lr = tid >> 2, lseg = tid & 3;
    const int jc = (n0 + lr < FDIM) ? (n0 + lr) : (FDIM - 1);   // clamped B row

    g2_issue(H, W2T, As[0], Bts[0], Bss[0], m0, 0,  jc, tid, lr, lseg);
    g2_issue(H, W2T, As[1], Bts[1], Bss[1], m0, TK, jc, tid, lr, lseg);

    for (int kt = 0; kt < KT; ++kt) {
        const int p = kt & 1;
        if (kt + 1 < KT) wait_async_le4();
        else             wait_async_le0();
        __syncthreads();

        Frag a[2], btf[2], bsf[2];
        #pragma unroll
        for (int mi = 0; mi < 2; ++mi) {
            const unsigned short* pa = &As[p][(wm * 32 + mi * 16 + lo) * LDT];
            a[mi].q[0] = *(const uint4*)(pa + hi * 8);
            a[mi].q[1] = *(const uint4*)(pa + 16 + hi * 8);
        }
        #pragma unroll
        for (int ni = 0; ni < 2; ++ni) {
            const unsigned short* pt = &Bts[p][(wn * 32 + ni * 16 + lo) * LDT];
            const unsigned short* ps = &Bss[p][(wn * 32 + ni * 16 + lo) * LDT];
            btf[ni].q[0] = *(const uint4*)(pt + hi * 8);
            btf[ni].q[1] = *(const uint4*)(pt + 16 + hi * 8);
            bsf[ni].q[0] = *(const uint4*)(ps + hi * 8);
            bsf[ni].q[1] = *(const uint4*)(ps + 16 + hi * 8);
        }
        #pragma unroll
        for (int mi = 0; mi < 2; ++mi)
            #pragma unroll
            for (int ni = 0; ni < 2; ++ni) {
                acct[mi][ni] = __builtin_amdgcn_wmma_f32_16x16x32_bf16(
                    false, a[mi].v, false, btf[ni].v, (short)0, acct[mi][ni], false, false);
                accs[mi][ni] = __builtin_amdgcn_wmma_f32_16x16x32_bf16(
                    false, a[mi].v, false, bsf[ni].v, (short)0, accs[mi][ni], false, false);
            }
        __syncthreads();                              // all waves done reading buf p

        if (kt + 2 < KT)
            g2_issue(H, W2T, As[p], Bts[p], Bss[p], m0, (kt + 2) * TK, jc, tid, lr, lseg);
    }

    // fused epilogue: t,s -> coupling -> batchnorm, ILDJ partials per row
    float g[2][8];
    #pragma unroll
    for (int mi = 0; mi < 2; ++mi)
        #pragma unroll
        for (int v = 0; v < 8; ++v) g[mi][v] = 0.f;

    #pragma unroll
    for (int mi = 0; mi < 2; ++mi)
        #pragma unroll
        for (int ni = 0; ni < 2; ++ni) {
            int j = n0 + wn * 32 + ni * 16 + lo;
            if (j < FDIM) {
                float b2t = b2[j], b2s = b2[FDIM + j];
                float mf  = (float)mask[j], im = 1.f - mf;
                float scale = __expf(lg[j]) * rsqrtf(rv[j] + EPSV);
                float rmj = rm[j], btj = bt[j];
                #pragma unroll
                for (int v = 0; v < 8; ++v) {
                    int r = m0 + wm * 32 + mi * 16 + v + hi * 8;
                    float t  = acct[mi][ni][v] + b2t;
                    float s  = tanhf(accs[mi][ni][v] + b2s);
                    float xv = curx[(size_t)r * FDIM + j];
                    float uu = mf * xv + im * ((xv - t) * __expf(-s));
                    g[mi][v] += im * (-s);
                    curx[(size_t)r * FDIM + j] = (uu - rmj) * scale + btj;
                }
            }
        }

    #pragma unroll
    for (int mi = 0; mi < 2; ++mi)
        #pragma unroll
        for (int v = 0; v < 8; ++v) {
            float p = g[mi][v];
            #pragma unroll
            for (int off = 1; off < 16; off <<= 1)    // reduce within 16-lane half
                p += __shfl_xor(p, off, 32);
            if (lo == 0) {
                int r = m0 + wm * 32 + mi * 16 + v + hi * 8;
                atomicAdd(&ildj[r], p);
            }
        }
}

// ---------------------------------------------------------------- final: prior + ildj + bn const
__global__ __launch_bounds__(256)
void final_kernel(const float* __restrict__ curx, const float* __restrict__ ildj,
                  const float* __restrict__ bnc, float* __restrict__ out) {
    int wave = threadIdx.x >> 5, lane = threadIdx.x & 31;
    int row  = blockIdx.x * 8 + wave;
    const float* xr = curx + (size_t)row * FDIM;
    float s = 0.f;
    for (int j = lane; j < FDIM; j += 32) { float x = xr[j]; s -= 0.5f * x * x; }
    #pragma unroll
    for (int off = 16; off > 0; off >>= 1) s += __shfl_xor(s, off, 32);
    if (lane == 0) out[row] = s - 0.5f * LOG2PI * (float)FDIM + ildj[row] + bnc[0];
}

// ---------------------------------------------------------------- launch
extern "C" void kernel_launch(void* const* d_in, const int* in_sizes, int n_in,
                              void* d_out, int out_size, void* d_ws, size_t ws_size,
                              hipStream_t stream) {
    (void)in_sizes; (void)n_in; (void)out_size; (void)ws_size;
    const float* x   = (const float*)d_in[0];
    const float* W1  = (const float*)d_in[1];
    const float* b1  = (const float*)d_in[2];
    const float* W2  = (const float*)d_in[3];
    const float* b2  = (const float*)d_in[4];
    const float* lg  = (const float*)d_in[5];
    const float* bt  = (const float*)d_in[6];
    const float* rm  = (const float*)d_in[7];
    const float* rv  = (const float*)d_in[8];
    const int*   msk = (const int*)d_in[9];
    float* out = (float*)d_out;

    char* ws = (char*)d_ws;
    size_t off = 0;
    float*          curx = (float*)(ws + off);          off += (size_t)BATCH * FDIM * 4;
    unsigned short* mx   = (unsigned short*)(ws + off); off += (size_t)BATCH * FDIM * 2;
    unsigned short* h    = (unsigned short*)(ws + off); off += (size_t)BATCH * UDIM * 2;
    unsigned short* w1T  = (unsigned short*)(ws + off); off += (size_t)UDIM * FDIM * 2;
    unsigned short* w2T  = (unsigned short*)(ws + off); off += (size_t)2 * FDIM * UDIM * 2;
    float*          ildj = (float*)(ws + off);          off += (size_t)BATCH * 4;
    float*          bnc  = (float*)(ws + off);          off += 256;

    zero_kernel<<<(BATCH + 255) / 256, 256, 0, stream>>>(ildj, BATCH);
    bnc_kernel<<<1, 256, 0, stream>>>(lg, rv, bnc);

    for (int i = 0; i < NFLOWS; ++i) {
        const float* src    = (i == 0) ? x : curx;
        const int*   mask_i = msk + i * FDIM;
        prep_kernel<<<(BATCH * FDIM + 255) / 256, 256, 0, stream>>>(src, mask_i, curx, mx);
        convT_kernel<<<(FDIM * UDIM + 255) / 256, 256, 0, stream>>>(
            W1 + (size_t)i * FDIM * UDIM, w1T, FDIM, UDIM);
        convT_kernel<<<(UDIM * 2 * FDIM + 255) / 256, 256, 0, stream>>>(
            W2 + (size_t)i * UDIM * 2 * FDIM, w2T, UDIM, 2 * FDIM);

        dim3 g1(UDIM / 128, BATCH / TM);
        gemm1_kernel<<<g1, 256, 0, stream>>>(mx, w1T, b1 + (size_t)i * UDIM, h);

        dim3 g2((FDIM + 63) / 64, BATCH / TM);
        gemm2_kernel<<<g2, 256, 0, stream>>>(h, w2T, b2 + (size_t)i * 2 * FDIM, mask_i,
                                             lg + (size_t)i * FDIM, bt + (size_t)i * FDIM,
                                             rm + (size_t)i * FDIM, rv + (size_t)i * FDIM,
                                             curx, ildj);
    }
    final_kernel<<<BATCH / 8, 256, 0, stream>>>(curx, ildj, bnc, out);
}
